// LongEncoderModule_69939247448730
// MI455X (gfx1250) — matrix operand order repaired and checked by reading, not compiled, not run on hardware
//
#include <hip/hip_runtime.h>
#include <hip/hip_bf16.h>
#include <math.h>

#define VOCAB 50265
#define SEQ 2048
#define EMB 768
#define NLAY 12
#define NHD 12
#define DHD 64
#define FFI 3072
#define CHK 256
#define NCH (SEQ / CHK)        /* 8 */
#define SPAD (SEQ + 2 * CHK)   /* 2560 */
#define NEGV (-1000000000.0f)

typedef __attribute__((ext_vector_type(16))) __bf16 bf16x16;
typedef __attribute__((ext_vector_type(8)))  float  floatx8;

__device__ __forceinline__ float bf2f(unsigned short h) {
  return __uint_as_float(((unsigned int)h) << 16);
}
__device__ __forceinline__ unsigned short f2bf(float f) {
  unsigned int u = __float_as_uint(f);
  u += 0x7fffu + ((u >> 16) & 1u);   // round-to-nearest-even
  return (unsigned short)(u >> 16);
}

struct FragBF { union { bf16x16 v; uint4 u[2]; }; };

// CDNA5 async direct global->LDS copy (16B per lane), tracked by ASYNCcnt.
// LDS operand is the wave-relative LDS byte address = low 32 bits of the
// generic pointer (flat LDS aperture maps addr[31:0]).
__device__ __forceinline__ void async_ld_b128(void* lds, const void* gmem) {
  unsigned loff = (unsigned)(size_t)lds;
  asm volatile("global_load_async_to_lds_b128 %0, %1, off"
               :: "v"(loff), "v"(gmem) : "memory");
}
__device__ __forceinline__ void wait_async0() {
  asm volatile("s_wait_asynccnt 0x0" ::: "memory");
}

// A-matrix 16x32 bf16 fragment (per cdna5 layout): lane = M (mod 16),
// lane>=16 selects K-halves 8..15 / 24..31. Elements 0..7 -> K=kb..kb+7,
// elements 8..15 -> K=kb+16..kb+23 (both contiguous -> two b128 LDS loads).
__device__ __forceinline__ FragBF frag_a(const unsigned short* base, int ld, int lane) {
  FragBF f;
  const unsigned short* p = base + (lane & 15) * ld + ((lane >> 4) * 8);
  f.u[0] = *(const uint4*)(p);
  f.u[1] = *(const uint4*)(p + 16);
  return f;
}
// B-matrix 32x16 bf16 fragment: lane = N (mod 16), same K order as frag_a:
// elements 0..7 -> K=8h..8h+7, elements 8..15 -> K=16+8h..16+8h+7 (h = lane>>4).
__device__ __forceinline__ FragBF frag_b(const unsigned short* base, int ld, int lane) {
  FragBF f;
  const unsigned short* p = base + (lane & 15) * ld + ((lane >> 4) * 8);
  f.u[0] = *(const uint4*)(p);
  f.u[1] = *(const uint4*)(p + 16);
  return f;
}
__device__ __forceinline__ floatx8 wmma_bf(const FragBF& a, const FragBF& b, floatx8 c) {
  return __builtin_amdgcn_wmma_f32_16x16x32_bf16(false, a.v, false, b.v,
                                                 (short)0, c, false, false);
}
__device__ __forceinline__ floatx8 zero8() {
  floatx8 z = {0.f, 0.f, 0.f, 0.f, 0.f, 0.f, 0.f, 0.f};
  return z;
}
__device__ __forceinline__ float gelu_exact(float x) {
  return 0.5f * x * (1.0f + erff(x * 0.70710678118654752f));
}
__device__ __forceinline__ float block_sum(float v, float* red, int tid) {
  red[tid] = v; __syncthreads();
  for (int st = 128; st > 0; st >>= 1) {
    if (tid < st) red[tid] += red[tid + st];
    __syncthreads();
  }
  float r = red[0]; __syncthreads();
  return r;
}
__device__ __forceinline__ float block_max(float v, float* red, int tid) {
  red[tid] = v; __syncthreads();
  for (int st = 128; st > 0; st >>= 1) {
    if (tid < st) red[tid] = fmaxf(red[tid], red[tid + st]);
    __syncthreads();
  }
  float r = red[0]; __syncthreads();
  return r;
}

// ---------------------------------------------------------------------------
// Weight fp32 [K][N] -> bf16 transposed [N][K] (tiled 32x32 via LDS)
// ---------------------------------------------------------------------------
__global__ void k_cvtT(const float* __restrict__ W, unsigned short* __restrict__ Wt,
                       int K, int N) {
  __shared__ float t[32][33];
  int bx = blockIdx.x * 32, by = blockIdx.y * 32;
  int tx = threadIdx.x, ty = threadIdx.y;   // 32 x 8
#pragma unroll
  for (int i = 0; i < 4; i++) {
    int k = by + ty + i * 8;
    t[ty + i * 8][tx] = W[(size_t)k * N + bx + tx];
  }
  __syncthreads();
#pragma unroll
  for (int i = 0; i < 4; i++) {
    int n = bx + ty + i * 8;
    Wt[(size_t)n * K + by + tx] = f2bf(t[tx][ty + i * 8]);
  }
}

__global__ void k_cvt(const float* __restrict__ a, unsigned short* __restrict__ b, int n) {
  size_t i = (size_t)blockIdx.x * 256 + threadIdx.x;
  if (i < (size_t)n) b[i] = f2bf(a[i]);
}

// ---------------------------------------------------------------------------
// Generic bf16 WMMA GEMM: C[M][N] = act(A[M][K] @ Wt[N][K]^T + bias)
// Block tile 128x64, 8 waves of 32x32 (2x2 wmma), K stepped 32.
// LDS double-buffered, filled via global_load_async_to_lds_b128 (ASYNCcnt).
// act: 0 = none, 1 = exact GELU.
// ---------------------------------------------------------------------------
#define GBM 128
#define GBN 64
#define GBK 32
__global__ __launch_bounds__(256) void k_gemm(
    const unsigned short* __restrict__ A, int lda,
    const unsigned short* __restrict__ Bt, int ldb,
    const float* __restrict__ bias,
    float* __restrict__ Cf, unsigned short* __restrict__ Cb, int ldc,
    int M, int N, int K, int act) {
  __shared__ __align__(16) unsigned short As[2][GBM * GBK];
  __shared__ __align__(16) unsigned short Bs[2][GBN * GBK];
  int tid = threadIdx.x, lane = tid & 31, wid = tid >> 5;
  int wm = wid & 3, wn = wid >> 2;                 // 4 waves in M, 2 in N
  int m0 = blockIdx.y * GBM, n0 = blockIdx.x * GBN;

  // per-thread staging slots: A = 512 x b128/stage (2 per thread), B = 256 (1)
  int aIdx0 = tid * 2, aIdx1 = tid * 2 + 1;
  int ar0 = aIdx0 >> 2, ac0 = (aIdx0 & 3) * 8;
  int ar1 = aIdx1 >> 2, ac1 = (aIdx1 & 3) * 8;
  int br = tid >> 2,   bc = (tid & 3) * 8;

  auto issueStage = [&](int buf, int kk) {
    async_ld_b128(&As[buf][ar0 * GBK + ac0], A + (size_t)(m0 + ar0) * lda + kk + ac0);
    async_ld_b128(&As[buf][ar1 * GBK + ac1], A + (size_t)(m0 + ar1) * lda + kk + ac1);
    async_ld_b128(&Bs[buf][br * GBK + bc],  Bt + (size_t)(n0 + br) * ldb + kk + bc);
  };

  issueStage(0, 0);

  floatx8 acc[2][2];
#pragma unroll
  for (int i = 0; i < 2; i++)
#pragma unroll
    for (int j = 0; j < 2; j++) acc[i][j] = zero8();

  wait_async0();
  __syncthreads();

  int nk = K / GBK;
  for (int kt = 0; kt < nk; ++kt) {
    int cur = kt & 1;
    if (kt + 1 < nk) issueStage(cur ^ 1, (kt + 1) * GBK);
    FragBF a0 = frag_a(&As[cur][(wm * 32 + 0) * GBK], GBK, lane);
    FragBF a1 = frag_a(&As[cur][(wm * 32 + 16) * GBK], GBK, lane);
    FragBF b0 = frag_b(&Bs[cur][(wn * 32 + 0) * GBK], GBK, lane);
    FragBF b1 = frag_b(&Bs[cur][(wn * 32 + 16) * GBK], GBK, lane);
    acc[0][0] = wmma_bf(a0, b0, acc[0][0]);
    acc[0][1] = wmma_bf(a0, b1, acc[0][1]);
    acc[1][0] = wmma_bf(a1, b0, acc[1][0]);
    acc[1][1] = wmma_bf(a1, b1, acc[1][1]);
    if (kt + 1 < nk) wait_async0();   // LDS writes of next stage complete
    __syncthreads();                  // publish to all waves
  }

#pragma unroll
  for (int i = 0; i < 2; i++) {
    int mt = m0 + wm * 32 + i * 16;
#pragma unroll
    for (int j = 0; j < 2; j++) {
      int nt = n0 + wn * 32 + j * 16;
      int col = nt + (lane & 15);
      int mb = mt + ((lane >> 4) * 8);
      float bc2 = bias ? bias[col] : 0.0f;
#pragma unroll
      for (int r = 0; r < 8; r++) {
        int m = mb + r;
        float v = acc[i][j][r] + bc2;
        if (act == 1) v = gelu_exact(v);
        if (Cf) Cf[(size_t)m * ldc + col] = v;
        if (Cb) Cb[(size_t)m * ldc + col] = f2bf(v);
      }
    }
  }
}

// ---------------------------------------------------------------------------
// kvm (padded key validity) = mask * (1-gmask), zero in pad regions
// ---------------------------------------------------------------------------
__global__ void k_kvm(const int* __restrict__ mask, const int* __restrict__ gmask,
                      int* __restrict__ kvm) {
  int p = blockIdx.x * 256 + threadIdx.x;
  if (p >= SPAD) return;
  int t = p - CHK;
  kvm[p] = (t >= 0 && t < SEQ) ? (mask[t] * (1 - gmask[t])) : 0;
}

// ---------------------------------------------------------------------------
// q/k/v [S][E] fp32 -> head-major bf16: qh[NH][S][DH] (scaled 1/8),
// kh/vh padded [NH][SPAD][DH], vT [NH][DH][SPAD]
// ---------------------------------------------------------------------------
__global__ __launch_bounds__(256) void k_reshape(
    const float* __restrict__ qf, const float* __restrict__ kf, const float* __restrict__ vf,
    unsigned short* __restrict__ qh, unsigned short* __restrict__ kh,
    unsigned short* __restrict__ vh, unsigned short* __restrict__ vT) {
  size_t gid = (size_t)blockIdx.x * 256 + threadIdx.x;
  int d = (int)(gid & 63);
  int p = (int)((gid >> 6) % SPAD);
  int h = (int)(gid / ((size_t)64 * SPAD));
  int t = p - CHK;
  bool in = (t >= 0 && t < SEQ);
  float kv = in ? kf[(size_t)t * EMB + h * DHD + d] : 0.f;
  float vv = in ? vf[(size_t)t * EMB + h * DHD + d] : 0.f;
  kh[((size_t)h * SPAD + p) * DHD + d] = f2bf(kv);
  vh[((size_t)h * SPAD + p) * DHD + d] = f2bf(vv);
  vT[((size_t)h * DHD + d) * SPAD + p] = f2bf(vv);
  if (in)
    qh[((size_t)h * SEQ + t) * DHD + d] = f2bf(qf[(size_t)t * EMB + h * DHD + d] * 0.125f);
}

// ---------------------------------------------------------------------------
// Windowed scores: per (head, chunk): s[256][768] = q_chunk @ k_win^T (WMMA),
// masked by band & kvm. 8 waves over rows, loop 12 tiles of 64 keys.
// Staging via async global->LDS.
// ---------------------------------------------------------------------------
__global__ __launch_bounds__(256) void k_scores(
    const unsigned short* __restrict__ qh, const unsigned short* __restrict__ kh,
    const int* __restrict__ kvm, float* __restrict__ scores) {
  int h = blockIdx.x, n = blockIdx.y;
  int tid = threadIdx.x, lane = tid & 31, wid = tid >> 5;
  __shared__ __align__(16) unsigned short Qs[CHK * DHD];   // 32 KB
  __shared__ __align__(16) unsigned short Ks[64 * DHD];    // 8 KB
  const unsigned short* qp = qh + ((size_t)h * SEQ + n * CHK) * DHD;
  const unsigned short* kp = kh + ((size_t)h * SPAD + n * CHK) * DHD;
#pragma unroll
  for (int i = 0; i < 8; i++)
    async_ld_b128(Qs + tid * DHD + i * 8, qp + (size_t)tid * DHD + i * 8);
  float* sb = scores + (((size_t)h * NCH + n) * CHK) * (3 * CHK);
  for (int jt = 0; jt < 12; ++jt) {
    __syncthreads();                       // prior iter done reading Ks
#pragma unroll
    for (int i = 0; i < 2; i++) {
      int idx = tid * 2 + i;               // 0..511
      int key = idx >> 3, dc = (idx & 7) * 8;
      async_ld_b128(Ks + key * DHD + dc,
                    kp + (size_t)(jt * 64 + key) * DHD + dc);
    }
    wait_async0();                         // (covers Q asyncs on first iter)
    __syncthreads();
    floatx8 acc[2][4];
#pragma unroll
    for (int i = 0; i < 2; i++)
#pragma unroll
      for (int j = 0; j < 4; j++) acc[i][j] = zero8();
#pragma unroll
    for (int ks = 0; ks < DHD; ks += 32) {
      FragBF a0 = frag_a(Qs + (wid * 32 + 0) * DHD + ks, DHD, lane);
      FragBF a1 = frag_a(Qs + (wid * 32 + 16) * DHD + ks, DHD, lane);
#pragma unroll
      for (int j4 = 0; j4 < 4; j4++) {
        FragBF b = frag_b(Ks + (j4 * 16) * DHD + ks, DHD, lane);
        acc[0][j4] = wmma_bf(a0, b, acc[0][j4]);
        acc[1][j4] = wmma_bf(a1, b, acc[1][j4]);
      }
    }
#pragma unroll
    for (int i = 0; i < 2; i++) {
#pragma unroll
      for (int j4 = 0; j4 < 4; j4++) {
        int col = jt * 64 + j4 * 16 + (lane & 15);
        int rbase = wid * 32 + i * 16 + ((lane >> 4) * 8);
        int kvv = kvm[n * CHK + col];
#pragma unroll
        for (int r = 0; r < 8; r++) {
          int row = rbase + r;
          int rel = col - CHK - row;
          bool ok = (rel >= -CHK) && (rel <= CHK) && (kvv != 0);
          sb[(size_t)row * (3 * CHK) + col] = ok ? acc[i][j4][r] : NEGV;
        }
      }
    }
  }
}

// ---------------------------------------------------------------------------
// Softmax over [s_glob, s_loc(768)] per row -> pl bf16 [NH][S][768], pg f32
// ---------------------------------------------------------------------------
__global__ __launch_bounds__(256) void k_softmax(
    const float* __restrict__ scores, const unsigned short* __restrict__ qh,
    const unsigned short* __restrict__ kh, unsigned short* __restrict__ plb,
    float* __restrict__ pg) {
  int h = blockIdx.x, n = blockIdx.y, r = threadIdx.x;
  int t = n * CHK + r;
  const unsigned short* qr = qh + ((size_t)h * SEQ + t) * DHD;
  const unsigned short* k0 = kh + ((size_t)h * SPAD + CHK) * DHD;   // token 0
  float sg = 0.f;
#pragma unroll 8
  for (int d = 0; d < DHD; ++d) sg += bf2f(qr[d]) * bf2f(k0[d]);
  const float* sr = scores + (((size_t)h * NCH + n) * CHK + r) * (3 * CHK);
  float mx = sg;
  for (int j = 0; j < 3 * CHK; j++) mx = fmaxf(mx, sr[j]);
  float eg = expf(sg - mx), sum = eg;
  for (int j = 0; j < 3 * CHK; j++) sum += expf(sr[j] - mx);
  float inv = 1.0f / sum;
  unsigned short* pr = plb + ((size_t)h * SEQ + t) * (3 * CHK);
  for (int j = 0; j < 3 * CHK; j++) pr[j] = f2bf(expf(sr[j] - mx) * inv);
  pg[(size_t)h * SEQ + t] = eg * inv;
}

// ---------------------------------------------------------------------------
// PV: out[256][64] = pl @ v_win (WMMA, 12 K-tiles of 64) + pg * v0,
// written to attn[S][E] at column h*64. Staging via async global->LDS.
// ---------------------------------------------------------------------------
__global__ __launch_bounds__(256) void k_av(
    const unsigned short* __restrict__ plb, const unsigned short* __restrict__ vT,
    const float* __restrict__ pg, const unsigned short* __restrict__ vh,
    float* __restrict__ attn) {
  int h = blockIdx.x, n = blockIdx.y;
  int tid = threadIdx.x, lane = tid & 31, wid = tid >> 5;
  __shared__ __align__(16) unsigned short Ps[CHK * 64];   // 32 KB
  __shared__ __align__(16) unsigned short Vs[64 * 64];    // 8 KB
  floatx8 acc[2][4];
#pragma unroll
  for (int i = 0; i < 2; i++)
#pragma unroll
    for (int j = 0; j < 4; j++) acc[i][j] = zero8();
  const unsigned short* pp = plb + ((size_t)h * SEQ + n * CHK) * (3 * CHK);
  for (int jt = 0; jt < 12; ++jt) {
    __syncthreads();
#pragma unroll
    for (int i = 0; i < 8; i++)
      async_ld_b128(Ps + tid * 64 + i * 8,
                    pp + (size_t)tid * (3 * CHK) + jt * 64 + i * 8);
#pragma unroll
    for (int i = 0; i < 2; i++) {
      int idx = tid * 2 + i;
      int d = idx >> 3, kc = (idx & 7) * 8;
      async_ld_b128(Vs + d * 64 + kc,
                    vT + ((size_t)h * DHD + d) * SPAD + n * CHK + jt * 64 + kc);
    }
    wait_async0();
    __syncthreads();
#pragma unroll
    for (int ks = 0; ks < 64; ks += 32) {
      FragBF a0 = frag_a(Ps + (wid * 32 + 0) * 64 + ks, 64, lane);
      FragBF a1 = frag_a(Ps + (wid * 32 + 16) * 64 + ks, 64, lane);
#pragma unroll
      for (int j4 = 0; j4 < 4; j4++) {
        FragBF b = frag_b(Vs + (j4 * 16) * 64 + ks, 64, lane);
        acc[0][j4] = wmma_bf(a0, b, acc[0][j4]);
        acc[1][j4] = wmma_bf(a1, b, acc[1][j4]);
      }
    }
  }
#pragma unroll
  for (int i = 0; i < 2; i++) {
#pragma unroll
    for (int j4 = 0; j4 < 4; j4++) {
      int d = j4 * 16 + (lane & 15);
      float v0 = bf2f(vh[((size_t)h * SPAD + CHK) * DHD + d]);
      int rbase = wid * 32 + i * 16 + ((lane >> 4) * 8);
#pragma unroll
      for (int r = 0; r < 8; r++) {
        int row = rbase + r, tok = n * CHK + row;
        attn[(size_t)tok * EMB + h * DHD + d] =
            acc[i][j4][r] + pg[(size_t)h * SEQ + tok] * v0;
      }
    }
  }
}

// ---------------------------------------------------------------------------
// Global-token query projection: qg[h*64+d] = (x0 . Wqg[:,o] + bqg[o]) / 8
// ---------------------------------------------------------------------------
__global__ __launch_bounds__(256) void k_qg(const float* __restrict__ x0,
                                            const float* __restrict__ Wqg,
                                            const float* __restrict__ bqg,
                                            float* __restrict__ qg) {
  int o = blockIdx.x * 256 + threadIdx.x;
  if (o >= EMB) return;
  float a = bqg[o];
  for (int e = 0; e < EMB; e++) a += x0[e] * Wqg[(size_t)e * EMB + o];
  qg[o] = a * 0.125f;
}

// ---------------------------------------------------------------------------
// Global attention for token 0: per-head softmax(qg.kg^T) @ vg -> attn[0][:]
// ---------------------------------------------------------------------------
__global__ __launch_bounds__(256) void k_gattn(
    const float* __restrict__ qg, const float* __restrict__ kgf,
    const float* __restrict__ vgf, const int* __restrict__ mask,
    float* __restrict__ attn) {
  int h = blockIdx.x, tid = threadIdx.x;
  __shared__ float p[SEQ];
  __shared__ float red[256];
  const float* q = qg + h * DHD;
  float lmax = -1e30f;
  for (int s = tid; s < SEQ; s += 256) {
    const float* kr = kgf + (size_t)s * EMB + h * DHD;
    float d = 0.f;
#pragma unroll 8
    for (int i = 0; i < DHD; i++) d += q[i] * kr[i];
    if (!mask[s]) d = NEGV;
    p[s] = d;
    lmax = fmaxf(lmax, d);
  }
  float mx = block_max(lmax, red, tid);
  float ls = 0.f;
  for (int s = tid; s < SEQ; s += 256) { float e = expf(p[s] - mx); p[s] = e; ls += e; }
  float inv = 1.0f / block_sum(ls, red, tid);
  __syncthreads();
  if (tid < DHD) {
    float o = 0.f;
    for (int s = 0; s < SEQ; s++) o += p[s] * vgf[(size_t)s * EMB + h * DHD + tid];
    attn[h * DHD + tid] = o * inv;   // row 0 of attn
  }
}

// ---------------------------------------------------------------------------
// y = LN(a (+ b)) * s + bias  -> yf fp32, yb bf16
// ---------------------------------------------------------------------------
__global__ __launch_bounds__(256) void k_ln(
    const float* __restrict__ a, const float* __restrict__ b,
    const float* __restrict__ sc, const float* __restrict__ bi,
    float* __restrict__ yf, unsigned short* __restrict__ yb) {
  int t = blockIdx.x, tid = threadIdx.x;
  __shared__ float red[256];
  float v[3]; float ls = 0.f;
#pragma unroll
  for (int i = 0; i < 3; i++) {
    int e = tid + i * 256;
    float x = a[(size_t)t * EMB + e];
    if (b) x += b[(size_t)t * EMB + e];
    v[i] = x; ls += x;
  }
  float mean = block_sum(ls, red, tid) * (1.0f / EMB);
  float lv = 0.f;
#pragma unroll
  for (int i = 0; i < 3; i++) { float d = v[i] - mean; lv += d * d; }
  float var = block_sum(lv, red, tid) * (1.0f / EMB);
  float rstd = rsqrtf(var + 1e-5f);
#pragma unroll
  for (int i = 0; i < 3; i++) {
    int e = tid + i * 256;
    float y = (v[i] - mean) * rstd * sc[e] + bi[e];
    yf[(size_t)t * EMB + e] = y;
    if (yb) yb[(size_t)t * EMB + e] = f2bf(y);
  }
}

// ---------------------------------------------------------------------------
// Embedding gather + LN
// ---------------------------------------------------------------------------
__global__ __launch_bounds__(256) void k_embed(
    const int* __restrict__ tok, const float* __restrict__ we,
    const float* __restrict__ pe, const float* __restrict__ sc,
    const float* __restrict__ bi, float* __restrict__ xf,
    unsigned short* __restrict__ xb) {
  int t = blockIdx.x, tid = threadIdx.x;
  __shared__ float red[256];
  int tk = tok[t];
  float v[3]; float ls = 0.f;
#pragma unroll
  for (int i = 0; i < 3; i++) {
    int e = tid + i * 256;
    float x = we[(size_t)tk * EMB + e] + pe[(size_t)(t + 2) * EMB + e];
    v[i] = x; ls += x;
  }
  float mean = block_sum(ls, red, tid) * (1.0f / EMB);
  float lv = 0.f;
#pragma unroll
  for (int i = 0; i < 3; i++) { float d = v[i] - mean; lv += d * d; }
  float var = block_sum(lv, red, tid) * (1.0f / EMB);
  float rstd = rsqrtf(var + 1e-5f);
#pragma unroll
  for (int i = 0; i < 3; i++) {
    int e = tid + i * 256;
    float y = (v[i] - mean) * rstd * sc[e] + bi[e];
    xf[(size_t)t * EMB + e] = y;
    xb[(size_t)t * EMB + e] = f2bf(y);
  }
}

// ---------------------------------------------------------------------------
// Classifier: logits [S][3] then argmax as float, concatenated in d_out
// ---------------------------------------------------------------------------
__global__ __launch_bounds__(256) void k_cls(
    const float* __restrict__ xf, const float* __restrict__ cw,
    const float* __restrict__ cb, float* __restrict__ out) {
  int t = blockIdx.x * 256 + threadIdx.x;
  if (t >= SEQ) return;
  float l0 = cb[0], l1 = cb[1], l2 = cb[2];
  for (int e = 0; e < EMB; e++) {
    float x = xf[(size_t)t * EMB + e];
    l0 += x * cw[e * 3 + 0];
    l1 += x * cw[e * 3 + 1];
    l2 += x * cw[e * 3 + 2];
  }
  out[t * 3 + 0] = l0; out[t * 3 + 1] = l1; out[t * 3 + 2] = l2;
  int best = 0; float bv = l0;
  if (l1 > bv) { bv = l1; best = 1; }
  if (l2 > bv) { bv = l2; best = 2; }
  out[SEQ * 3 + t] = (float)best;
}

// ---------------------------------------------------------------------------
// Host orchestration
// ---------------------------------------------------------------------------
static void run_cvtT(const float* W, unsigned short* Wt, int K, int N, hipStream_t s) {
  dim3 g(N / 32, K / 32), b(32, 8);
  k_cvtT<<<g, b, 0, s>>>(W, Wt, K, N);
}
static void run_gemm(const unsigned short* A, const unsigned short* Bt,
                     const float* bias, float* Cf, unsigned short* Cb,
                     int M, int N, int K, int act, hipStream_t s) {
  dim3 g(N / GBN, M / GBM);
  k_gemm<<<g, 256, 0, s>>>(A, K, Bt, K, bias, Cf, Cb, N, M, N, K, act);
}

extern "C" void kernel_launch(void* const* d_in, const int* in_sizes, int n_in,
                              void* d_out, int out_size, void* d_ws, size_t ws_size,
                              hipStream_t stream) {
  (void)in_sizes; (void)n_in; (void)out_size; (void)ws_size;
  const int*   tok    = (const int*)d_in[0];
  const int*   mask   = (const int*)d_in[1];
  const int*   gmask  = (const int*)d_in[2];
  const float* wemb   = (const float*)d_in[3];
  const float* pemb   = (const float*)d_in[4];
  const float* embs   = (const float*)d_in[5];
  const float* embb   = (const float*)d_in[6];
  const float* Wq = (const float*)d_in[7],  *Wk = (const float*)d_in[8];
  const float* Wv = (const float*)d_in[9],  *Wqg = (const float*)d_in[10];
  const float* Wkg = (const float*)d_in[11], *Wvg = (const float*)d_in[12];
  const float* Wo = (const float*)d_in[13];
  const float* bq = (const float*)d_in[14], *bk = (const float*)d_in[15];
  const float* bv = (const float*)d_in[16], *bqg = (const float*)d_in[17];
  const float* bkg = (const float*)d_in[18], *bvg = (const float*)d_in[19];
  const float* bo = (const float*)d_in[20];
  const float* W1 = (const float*)d_in[21], *b1 = (const float*)d_in[22];
  const float* W2 = (const float*)d_in[23], *b2 = (const float*)d_in[24];
  const float* ln1s = (const float*)d_in[25], *ln1b = (const float*)d_in[26];
  const float* ln2s = (const float*)d_in[27], *ln2b = (const float*)d_in[28];
  const float* clsW = (const float*)d_in[29], *clsb = (const float*)d_in[30];

  char* wp = (char*)d_ws;
  auto take = [&](size_t bytes) -> char* {
    char* r = wp;
    wp += (bytes + 255) & ~(size_t)255;
    return r;
  };
  const size_t SE = (size_t)SEQ * EMB;
  float* xf   = (float*)take(SE * 4);
  float* x1f  = (float*)take(SE * 4);
  float* qf   = (float*)take(SE * 4);
  float* kf   = (float*)take(SE * 4);
  float* vf   = (float*)take(SE * 4);
  float* kgf  = (float*)take(SE * 4);
  float* vgf  = (float*)take(SE * 4);
  float* attnf = (float*)take(SE * 4);
  float* aof  = (float*)take(SE * 4);
  float* ff   = (float*)take(SE * 4);
  unsigned short* xb    = (unsigned short*)take(SE * 2);
  unsigned short* x1b   = (unsigned short*)take(SE * 2);
  unsigned short* attnb = (unsigned short*)take(SE * 2);
  unsigned short* qh    = (unsigned short*)take(SE * 2);
  unsigned short* kh    = (unsigned short*)take((size_t)NHD * SPAD * DHD * 2);
  unsigned short* vh    = (unsigned short*)take((size_t)NHD * SPAD * DHD * 2);
  unsigned short* vT    = (unsigned short*)take((size_t)NHD * SPAD * DHD * 2);
  float* scores = (float*)take((size_t)NHD * NCH * CHK * (3 * CHK) * 4);
  unsigned short* plb = (unsigned short*)take((size_t)NHD * SEQ * (3 * CHK) * 2);
  float* pg = (float*)take((size_t)NHD * SEQ * 4);
  float* qg = (float*)take((size_t)EMB * 4);
  unsigned short* hb = (unsigned short*)take((size_t)SEQ * FFI * 2);
  unsigned short* wt = (unsigned short*)take((size_t)EMB * FFI * 2);
  int* kvm = (int*)take((size_t)SPAD * 4);

  // masks + embedding (once)
  k_kvm<<<(SPAD + 255) / 256, 256, 0, stream>>>(mask, gmask, kvm);
  k_embed<<<SEQ, 256, 0, stream>>>(tok, wemb, pemb, embs, embb, xf, xb);

  const size_t EE = (size_t)EMB * EMB;
  for (int l = 0; l < NLAY; ++l) {
    const float* Wq_l = Wq + (size_t)l * EE;
    const float* Wk_l = Wk + (size_t)l * EE;
    const float* Wv_l = Wv + (size_t)l * EE;
    const float* Wqg_l = Wqg + (size_t)l * EE;
    const float* Wkg_l = Wkg + (size_t)l * EE;
    const float* Wvg_l = Wvg + (size_t)l * EE;
    const float* Wo_l = Wo + (size_t)l * EE;
    const float* W1_l = W1 + (size_t)l * EMB * FFI;
    const float* W2_l = W2 + (size_t)l * FFI * EMB;

    // --- q/k/v projections (WMMA GEMM) ---
    run_cvtT(Wq_l, wt, EMB, EMB, stream);
    run_gemm(xb, wt, bq + (size_t)l * EMB, qf, nullptr, SEQ, EMB, EMB, 0, stream);
    run_cvtT(Wk_l, wt, EMB, EMB, stream);
    run_gemm(xb, wt, bk + (size_t)l * EMB, kf, nullptr, SEQ, EMB, EMB, 0, stream);
    run_cvtT(Wv_l, wt, EMB, EMB, stream);
    run_gemm(xb, wt, bv + (size_t)l * EMB, vf, nullptr, SEQ, EMB, EMB, 0, stream);
    k_reshape<<<(int)(((size_t)NHD * SPAD * DHD) / 256), 256, 0, stream>>>(
        qf, kf, vf, qh, kh, vh, vT);

    // --- global-attention projections ---
    run_cvtT(Wkg_l, wt, EMB, EMB, stream);
    run_gemm(xb, wt, bkg + (size_t)l * EMB, kgf, nullptr, SEQ, EMB, EMB, 0, stream);
    run_cvtT(Wvg_l, wt, EMB, EMB, stream);
    run_gemm(xb, wt, bvg + (size_t)l * EMB, vgf, nullptr, SEQ, EMB, EMB, 0, stream);
    k_qg<<<3, 256, 0, stream>>>(xf, Wqg_l, bqg + (size_t)l * EMB, qg);

    // --- windowed attention (WMMA) ---
    k_scores<<<dim3(NHD, NCH), 256, 0, stream>>>(qh, kh, kvm, scores);
    k_softmax<<<dim3(NHD, NCH), 256, 0, stream>>>(scores, qh, kh, plb, pg);
    k_av<<<dim3(NHD, NCH), 256, 0, stream>>>(plb, vT, pg, vh, attnf);
    k_gattn<<<NHD, 256, 0, stream>>>(qg, kgf, vgf, mask, attnf);  // overwrites row 0

    // --- output projection + LN1 ---
    k_cvt<<<(int)(SE / 256), 256, 0, stream>>>(attnf, attnb, (int)SE);
    run_cvtT(Wo_l, wt, EMB, EMB, stream);
    run_gemm(attnb, wt, bo + (size_t)l * EMB, aof, nullptr, SEQ, EMB, EMB, 0, stream);
    k_ln<<<SEQ, 256, 0, stream>>>(xf, aof, ln1s + (size_t)l * EMB,
                                  ln1b + (size_t)l * EMB, x1f, x1b);

    // --- FFN (WMMA, fused exact GELU) + LN2 ---
    run_cvtT(W1_l, wt, EMB, FFI, stream);
    run_gemm(x1b, wt, b1 + (size_t)l * FFI, nullptr, hb, SEQ, FFI, EMB, 1, stream);
    run_cvtT(W2_l, wt, FFI, EMB, stream);
    run_gemm(hb, wt, b2 + (size_t)l * EMB, ff, nullptr, SEQ, EMB, FFI, 0, stream);
    k_ln<<<SEQ, 256, 0, stream>>>(x1f, ff, ln2s + (size_t)l * EMB,
                                  ln2b + (size_t)l * EMB, xf, xb);
  }

  k_cls<<<SEQ / 256, 256, 0, stream>>>(xf, clsW, clsb, (float*)d_out);
}
